// Transformer_28441273434658
// MI455X (gfx1250) — matrix operand order repaired and checked by reading, not compiled, not run on hardware
//
#include <hip/hip_runtime.h>
#include <hip/hip_bf16.h>

// ---------------------------------------------------------------------------
// Transformer forward for MI455X (gfx1250, wave32, WMMA 16x16x32 f16,
// async global->LDS staging with ASYNCcnt double buffering)
// ---------------------------------------------------------------------------

#define BB   2
#define SS   1024
#define DD   1024
#define HH   16
#define HD   64
#define NL   6
#define DFF  4096
#define MM   (BB * SS)   // 2048 rows

typedef __attribute__((ext_vector_type(16))) _Float16 v16h;
typedef __attribute__((ext_vector_type(8)))  _Float16 v8h;
typedef __attribute__((ext_vector_type(8)))  float    v8f;

union hv16 { v16h v; v8h h8[2]; };

__device__ __forceinline__ v8f wmma_f16(const hv16& a, const hv16& b, v8f c) {
    return __builtin_amdgcn_wmma_f32_16x16x32_f16(
        /*neg_a=*/false, a.v, /*neg_b=*/false, b.v,
        /*c_mod=*/(short)0, c, /*reuse_a=*/false, /*reuse_b=*/false);
}

__device__ __forceinline__ float gelu_exact(float x) {
    return 0.5f * x * (1.0f + erff(x * 0.70710678118654752f));
}

// Async copy 16 bytes per lane: global memory -> LDS, tracked by ASYNCcnt.
// Generic pointers to __shared__ carry the LDS byte address in the low 32 bits.
__device__ __forceinline__ void async_copy_b128(const _Float16* g, _Float16* lds) {
    unsigned laddr = (unsigned)(unsigned long long)lds;
    asm volatile("global_load_async_to_lds_b128 %0, %1, off"
                 :
                 : "v"(laddr), "v"(g)
                 : "memory");
}

__device__ __forceinline__ void wait_asynccnt_le2() {
    asm volatile("s_wait_asynccnt 0x2" ::: "memory");
}

__device__ __forceinline__ void wait_asynccnt_0() {
    asm volatile("s_wait_asynccnt 0x0" ::: "memory");
}

// ---------------------------------------------------------------------------
// Weight convert + transpose: src [L][K][N] f32  ->  dst [L][N][K] f16
// ---------------------------------------------------------------------------
__global__ void cvt_transpose_kernel(const float* __restrict__ src,
                                     _Float16* __restrict__ dst,
                                     int Kd, int Nd) {
    long base = (long)blockIdx.z * (long)Kd * (long)Nd;
    int idx = blockIdx.x * 256 + threadIdx.x;
    if (idx >= Kd * Nd) return;
    int kk = idx / Nd;
    int nn = idx - kk * Nd;
    dst[base + (long)nn * Kd + kk] = (_Float16)src[base + idx];
}

// ---------------------------------------------------------------------------
// y = x + pos (broadcast over batch); emit f32 and f16 copies
// ---------------------------------------------------------------------------
__global__ void add_pos_kernel(const float* __restrict__ x,
                               const float* __restrict__ pos,
                               float* __restrict__ y,
                               _Float16* __restrict__ yh) {
    long i = (long)blockIdx.x * 256 + threadIdx.x;
    long sd = i % ((long)SS * DD);
    float v = x[i] + pos[sd];
    y[i]  = v;
    yh[i] = (_Float16)v;
}

// ---------------------------------------------------------------------------
// Generic WMMA GEMM:  C[M][N] = A[M][K] * Bt[N][K]^T + bias, optional GELU.
// Block tile 64x64, 256 threads (8 waves), wave tile 16x32 (2 accumulators).
// K tiles (32 wide) staged to LDS via async-to-LDS loads, double buffered.
// The K loop is peeled: steady state is unconditional
//   issue-next -> s_wait_asynccnt 2 -> barrier -> 2x WMMA -> barrier
// so the next tile's DMA overlaps the current tile's WMMAs with no branches.
// blockIdx.z offsets A / Bt / C pointers (used for per-head QKV projections).
// Requires K >= 64 (nk >= 2); true for all GEMMs here.
// ---------------------------------------------------------------------------
__global__ __launch_bounds__(256) void gemm_f16_kernel(
    const _Float16* __restrict__ A,  int lda, long a_off_z,
    const _Float16* __restrict__ Bt, int ldb, long b_off_z,
    const float* __restrict__ bias,
    float* __restrict__ Cf, _Float16* __restrict__ Ch, int ldc, long c_off_z,
    int M, int N, int K, int act)
{
    __shared__ __align__(16) _Float16 As[2][64][40];
    __shared__ __align__(16) _Float16 Bs[2][64][40];

    int z = blockIdx.z;
    A  += (long)z * a_off_z;
    Bt += (long)z * b_off_z;
    long coff = (long)z * c_off_z;

    int tid  = threadIdx.x;
    int wave = tid >> 5;
    int lane = tid & 31;
    int r    = lane & 15;
    bool hiL = lane >= 16;
    int wm   = wave & 3;    // 4 M sub-tiles of 16
    int wn   = wave >> 2;   // 2 N sub-tiles of 32

    int m0 = blockIdx.x * 64;
    int n0 = blockIdx.y * 64;

    int lrow = tid >> 2;          // 0..63
    int lcol = (tid & 3) * 8;     // 0,8,16,24

    const _Float16* gA = &A [(long)(m0 + lrow) * lda + lcol];
    const _Float16* gB = &Bt[(long)(n0 + lrow) * ldb + lcol];

    v8f acc0 = {};
    v8f acc1 = {};

    int nk = K >> 5;  // K / 32, >= 2

    // prologue: tile 0 -> buffer 0 (2 async ops per lane)
    async_copy_b128(gA, &As[0][lrow][lcol]);
    async_copy_b128(gB, &Bs[0][lrow][lcol]);

    // precomputed fragment source addresses (per buffer)
    const _Float16* aAddr0 = &As[0][wm * 16 + r][hiL ? 16 :  0];
    const _Float16* aAddr1 = &As[0][wm * 16 + r][hiL ? 24 :  8];
    const _Float16* b0Addr0 = &Bs[0][wn * 32 + r     ][hiL ? 16 : 0];
    const _Float16* b0Addr1 = &Bs[0][wn * 32 + r     ][hiL ? 24 : 8];
    const _Float16* b1Addr0 = &Bs[0][wn * 32 + 16 + r][hiL ? 16 : 0];
    const _Float16* b1Addr1 = &Bs[0][wn * 32 + 16 + r][hiL ? 24 : 8];
    const long bufA = (long)(&As[1][0][0] - &As[0][0][0]);   // halves
    const long bufB = (long)(&Bs[1][0][0] - &Bs[0][0][0]);

    // steady state: iterations 0 .. nk-2, unconditional prefetch of ki+1
    for (int ki = 0; ki < nk - 1; ++ki) {
        long ofsA = (ki & 1) ? bufA : 0;
        long ofsB = (ki & 1) ? bufB : 0;
        long nofA = (ki & 1) ? 0 : bufA;
        long nofB = (ki & 1) ? 0 : bufB;

        // prefetch next tile into the other buffer (safe: last read of that
        // buffer was before the trailing barrier of iteration ki-1)
        async_copy_b128(gA + (long)(ki + 1) * 32, (_Float16*)&As[0][lrow][lcol] + nofA);
        async_copy_b128(gB + (long)(ki + 1) * 32, (_Float16*)&Bs[0][lrow][lcol] + nofB);
        wait_asynccnt_le2();   // current tile's 2 copies complete (FIFO)
        __syncthreads();

        hv16 af, bf0, bf1;
        af.h8[0]  = *(const v8h*)(aAddr0 + ofsA);
        af.h8[1]  = *(const v8h*)(aAddr1 + ofsA);
        bf0.h8[0] = *(const v8h*)(b0Addr0 + ofsB);
        bf0.h8[1] = *(const v8h*)(b0Addr1 + ofsB);
        bf1.h8[0] = *(const v8h*)(b1Addr0 + ofsB);
        bf1.h8[1] = *(const v8h*)(b1Addr1 + ofsB);

        acc0 = wmma_f16(af, bf0, acc0);
        acc1 = wmma_f16(af, bf1, acc1);
        __syncthreads();
    }

    // epilogue: last tile (buffer (nk-1)&1)
    {
        long ofsA = ((nk - 1) & 1) ? bufA : 0;
        long ofsB = ((nk - 1) & 1) ? bufB : 0;
        wait_asynccnt_0();
        __syncthreads();

        hv16 af, bf0, bf1;
        af.h8[0]  = *(const v8h*)(aAddr0 + ofsA);
        af.h8[1]  = *(const v8h*)(aAddr1 + ofsA);
        bf0.h8[0] = *(const v8h*)(b0Addr0 + ofsB);
        bf0.h8[1] = *(const v8h*)(b0Addr1 + ofsB);
        bf1.h8[0] = *(const v8h*)(b1Addr0 + ofsB);
        bf1.h8[1] = *(const v8h*)(b1Addr1 + ofsB);

        acc0 = wmma_f16(af, bf0, acc0);
        acc1 = wmma_f16(af, bf1, acc1);
    }

    int col0 = n0 + wn * 32 + r;
    int col1 = col0 + 16;
    float b0 = bias[col0];
    float b1 = bias[col1];

#pragma unroll
    for (int j = 0; j < 8; ++j) {
        int row = m0 + wm * 16 + j + (hiL ? 8 : 0);
        float v0 = acc0[j] + b0;
        float v1 = acc1[j] + b1;
        if (act == 1) { v0 = gelu_exact(v0); v1 = gelu_exact(v1); }
        long i0 = coff + (long)row * ldc + col0;
        long i1 = coff + (long)row * ldc + col1;
        if (Cf) { Cf[i0] = v0; Cf[i1] = v1; }
        if (Ch) { Ch[i0] = (_Float16)v0; Ch[i1] = (_Float16)v1; }
    }
}

// ---------------------------------------------------------------------------
// Fused causal flash-attention. q/k/v/o are f16 [B,S,H*HD]; one wave owns a
// 16-row Q tile; K blocks of 32 columns; QK^T and P*V on WMMA; online softmax.
// ---------------------------------------------------------------------------
__global__ __launch_bounds__(256) void attn_flash_kernel(
    const _Float16* __restrict__ Q, const _Float16* __restrict__ Kh,
    const _Float16* __restrict__ V, _Float16* __restrict__ O, float scale)
{
    __shared__ __align__(16) _Float16 Psh[8][16][40];   // per-wave P transpose scratch

    int bh = blockIdx.y;
    int b  = bh / HH;
    int h  = bh - b * HH;
    int wave = threadIdx.x >> 5;
    int lane = threadIdx.x & 31;
    int r    = lane & 15;
    bool hiL = lane >= 16;

    int q0 = blockIdx.x * 128 + wave * 16;

    const _Float16* qb = Q + ((long)b * SS) * DD + h * HD;
    const _Float16* kb = Kh + ((long)b * SS) * DD + h * HD;
    const _Float16* vb = V + ((long)b * SS) * DD + h * HD;

    // Q fragments (M=16, K=64 -> two 16x32 A fragments), loaded once
    hv16 aq0, aq1;
    {
        const _Float16* qrow = qb + (long)(q0 + r) * DD;
        aq0.h8[0] = *(const v8h*)&qrow[hiL ? 16 :  0];
        aq0.h8[1] = *(const v8h*)&qrow[hiL ? 24 :  8];
        aq1.h8[0] = *(const v8h*)&qrow[32 + (hiL ? 16 :  0)];
        aq1.h8[1] = *(const v8h*)&qrow[32 + (hiL ? 24 :  8)];
    }

    float m[8], l[8];
    v8f o0 = {}, o1 = {}, o2 = {}, o3 = {};
#pragma unroll
    for (int j = 0; j < 8; ++j) { m[j] = -3.0e38f; l[j] = 0.0f; }

    int kmax = q0 + 15;   // causal upper bound (wave-uniform)

    for (int kb32 = 0; kb32 <= kmax; kb32 += 32) {
        // ---- scores for two adjacent 16-wide K tiles -----------------------
        v8f s0 = {}, s1 = {};
#pragma unroll
        for (int t = 0; t < 2; ++t) {
            int krow = kb32 + t * 16 + r;
            const _Float16* kr = kb + (long)krow * DD;
            hv16 bk0, bk1;
            bk0.h8[0] = *(const v8h*)&kr[hiL ? 16 : 0];
            bk0.h8[1] = *(const v8h*)&kr[hiL ? 24 : 8];
            bk1.h8[0] = *(const v8h*)&kr[32 + (hiL ? 16 : 0)];
            bk1.h8[1] = *(const v8h*)&kr[32 + (hiL ? 24 : 8)];
            v8f s = {};
            s = wmma_f16(aq0, bk0, s);
            s = wmma_f16(aq1, bk1, s);
            if (t == 0) s0 = s; else s1 = s;
        }

        // ---- mask (-1e20 pre-scale, as reference), online softmax ---------
#pragma unroll
        for (int j = 0; j < 8; ++j) {
            int qrow = q0 + j + (hiL ? 8 : 0);
            float x0 = ((kb32 + r)      <= qrow) ? s0[j] : -1.0e20f;
            float x1 = ((kb32 + 16 + r) <= qrow) ? s1[j] : -1.0e20f;
            x0 *= scale;
            x1 *= scale;
            float tmax = fmaxf(x0, x1);
            tmax = fmaxf(tmax, __shfl_xor(tmax, 1, 32));
            tmax = fmaxf(tmax, __shfl_xor(tmax, 2, 32));
            tmax = fmaxf(tmax, __shfl_xor(tmax, 4, 32));
            tmax = fmaxf(tmax, __shfl_xor(tmax, 8, 32));
            float mn    = fmaxf(m[j], tmax);
            float alpha = __expf(m[j] - mn);
            float p0 = __expf(x0 - mn);
            float p1 = __expf(x1 - mn);
            float ps = p0 + p1;
            ps += __shfl_xor(ps, 1, 32);
            ps += __shfl_xor(ps, 2, 32);
            ps += __shfl_xor(ps, 4, 32);
            ps += __shfl_xor(ps, 8, 32);
            l[j] = l[j] * alpha + ps;
            m[j] = mn;
            o0[j] *= alpha; o1[j] *= alpha; o2[j] *= alpha; o3[j] *= alpha;
            s0[j] = p0;
            s1[j] = p1;
        }

        // ---- P (16x32) -> LDS, re-read in A-fragment layout ---------------
#pragma unroll
        for (int j = 0; j < 8; ++j) {
            int row = j + (hiL ? 8 : 0);
            Psh[wave][row][r]      = (_Float16)s0[j];
            Psh[wave][row][16 + r] = (_Float16)s1[j];
        }
        // same-wave DS ops are in-order on CDNA5: no barrier required
        hv16 ap;
        ap.h8[0] = *(const v8h*)&Psh[wave][r][hiL ? 16 :  0];
        ap.h8[1] = *(const v8h*)&Psh[wave][r][hiL ? 24 :  8];

        // ---- O += P @ V_tile (four 16-col d tiles) ------------------------
#pragma unroll
        for (int f = 0; f < 4; ++f) {
            hv16 bv;
#pragma unroll
            for (int e = 0; e < 16; ++e) {
                int krow = kb32 + (hiL ? 16 : 0) + e;
                bv.v[e] = vb[(long)krow * DD + f * 16 + r];
            }
            if (f == 0)      o0 = wmma_f16(ap, bv, o0);
            else if (f == 1) o1 = wmma_f16(ap, bv, o1);
            else if (f == 2) o2 = wmma_f16(ap, bv, o2);
            else             o3 = wmma_f16(ap, bv, o3);
        }
    }

    // ---- normalize and store ---------------------------------------------
#pragma unroll
    for (int j = 0; j < 8; ++j) {
        float inv = 1.0f / l[j];
        int row = q0 + j + (hiL ? 8 : 0);
        _Float16* orow = O + ((long)b * SS + row) * DD + h * HD;
        orow[r]      = (_Float16)(o0[j] * inv);
        orow[16 + r] = (_Float16)(o1[j] * inv);
        orow[32 + r] = (_Float16)(o2[j] * inv);
        orow[48 + r] = (_Float16)(o3[j] * inv);
    }
}

// ---------------------------------------------------------------------------
// LayerNorm with optional residual: out = LN(A + Bres) * w + b
// One 256-thread block per row of D=1024.
// ---------------------------------------------------------------------------
__global__ __launch_bounds__(256) void ln_res_kernel(
    const float* __restrict__ A, const float* __restrict__ Bres,
    const float* __restrict__ w, const float* __restrict__ bias,
    float* __restrict__ outF, _Float16* __restrict__ outH)
{
    __shared__ float red[2][8];
    long row = blockIdx.x;
    const float* a  = A + row * DD;
    const float* br = Bres ? Bres + row * DD : nullptr;
    int tid = threadIdx.x;

    float z[4];
    float s = 0.0f, s2 = 0.0f;
#pragma unroll
    for (int i = 0; i < 4; ++i) {
        int c = tid + i * 256;
        float v = a[c] + (br ? br[c] : 0.0f);
        z[i] = v;
        s  += v;
        s2 += v * v;
    }
#pragma unroll
    for (int off = 16; off >= 1; off >>= 1) {
        s  += __shfl_xor(s,  off, 32);
        s2 += __shfl_xor(s2, off, 32);
    }
    int wave = tid >> 5, lane = tid & 31;
    if (lane == 0) { red[0][wave] = s; red[1][wave] = s2; }
    __syncthreads();
    if (wave == 0) {
        s  = (lane < 8) ? red[0][lane] : 0.0f;
        s2 = (lane < 8) ? red[1][lane] : 0.0f;
#pragma unroll
        for (int off = 4; off >= 1; off >>= 1) {
            s  += __shfl_xor(s,  off, 32);
            s2 += __shfl_xor(s2, off, 32);
        }
        if (lane == 0) { red[0][0] = s; red[1][0] = s2; }
    }
    __syncthreads();
    s  = red[0][0];
    s2 = red[1][0];
    float mean = s / (float)DD;
    float var  = s2 / (float)DD - mean * mean;
    float rstd = rsqrtf(var + 1e-5f);
#pragma unroll
    for (int i = 0; i < 4; ++i) {
        int c = tid + i * 256;
        float v = (z[i] - mean) * rstd * w[c] + bias[c];
        if (outF) outF[row * DD + c] = v;
        if (outH) outH[row * DD + c] = (_Float16)v;
    }
}

// ---------------------------------------------------------------------------
// Host-side orchestration
// ---------------------------------------------------------------------------
extern "C" void kernel_launch(void* const* d_in, const int* in_sizes, int n_in,
                              void* d_out, int out_size, void* d_ws, size_t ws_size,
                              hipStream_t stream) {
    const float* x     = (const float*)d_in[0];
    const float* pos   = (const float*)d_in[1];
    const float* Wq    = (const float*)d_in[2];
    const float* bq    = (const float*)d_in[3];
    const float* Wk    = (const float*)d_in[4];
    const float* bk    = (const float*)d_in[5];
    const float* Wv    = (const float*)d_in[6];
    const float* bv    = (const float*)d_in[7];
    const float* Wfc   = (const float*)d_in[8];
    const float* bfc   = (const float*)d_in[9];
    const float* ln1_w = (const float*)d_in[10];
    const float* ln1_b = (const float*)d_in[11];
    const float* ln2_w = (const float*)d_in[12];
    const float* ln2_b = (const float*)d_in[13];
    const float* W1    = (const float*)d_in[14];
    const float* b1    = (const float*)d_in[15];
    const float* W2    = (const float*)d_in[16];
    const float* b2    = (const float*)d_in[17];
    const float* lnf_w = (const float*)d_in[18];
    const float* lnf_b = (const float*)d_in[19];

    const long BSD  = (long)BB * SS * DD;   // 2,097,152
    const long BSF  = (long)BB * SS * DFF;  // 8,388,608

    char* ws = (char*)d_ws;
    size_t off = 0;
    auto alloc = [&](size_t bytes) -> char* {
        char* p = ws + off;
        off += (bytes + 255) & ~(size_t)255;
        return p;
    };

    float*    y_f32  = (float*)alloc(BSD * 4);
    float*    addf   = (float*)alloc(BSD * 4);     // residual 'add'
    float*    tmpf   = (float*)alloc(BSD * 4);     // gemm f32 outputs
    _Float16* y16    = (_Float16*)alloc(BSD * 2);
    _Float16* q16    = (_Float16*)alloc(BSD * 2);
    _Float16* k16    = (_Float16*)alloc(BSD * 2);
    _Float16* v16    = (_Float16*)alloc(BSD * 2);
    _Float16* ao16   = (_Float16*)alloc(BSD * 2);  // attention output (f16)
    _Float16* add16  = (_Float16*)alloc(BSD * 2);
    _Float16* ff16   = (_Float16*)alloc(BSF * 2);
    _Float16* WqT    = (_Float16*)alloc((size_t)NL * HD * HD * 2);
    _Float16* WkT    = (_Float16*)alloc((size_t)NL * HD * HD * 2);
    _Float16* WvT    = (_Float16*)alloc((size_t)NL * HD * HD * 2);
    _Float16* WfcT   = (_Float16*)alloc((size_t)NL * DD * DD * 2);
    _Float16* W1T    = (_Float16*)alloc((size_t)NL * DD * DFF * 2);
    _Float16* W2T    = (_Float16*)alloc((size_t)NL * DFF * DD * 2);

    // ---- weight conversion (every call; deterministic) --------------------
    {
        dim3 g((HD * HD + 255) / 256, 1, NL);
        cvt_transpose_kernel<<<g, 256, 0, stream>>>(Wq, WqT, HD, HD);
        cvt_transpose_kernel<<<g, 256, 0, stream>>>(Wk, WkT, HD, HD);
        cvt_transpose_kernel<<<g, 256, 0, stream>>>(Wv, WvT, HD, HD);
    }
    cvt_transpose_kernel<<<dim3((DD * DD + 255) / 256, 1, NL), 256, 0, stream>>>(Wfc, WfcT, DD, DD);
    cvt_transpose_kernel<<<dim3((DD * DFF + 255) / 256, 1, NL), 256, 0, stream>>>(W1, W1T, DD, DFF);
    cvt_transpose_kernel<<<dim3((DFF * DD + 255) / 256, 1, NL), 256, 0, stream>>>(W2, W2T, DFF, DD);

    // ---- embed ------------------------------------------------------------
    add_pos_kernel<<<(int)(BSD / 256), 256, 0, stream>>>(x, pos, y_f32, y16);

    const float scale = 0.125f;   // 1/sqrt(HD)

    for (int l = 0; l < NL; ++l) {
        // QKV per-head projections: M=2048, N=64, K=64, blockIdx.z = head
        dim3 gq(MM / 64, 1, HH);
        gemm_f16_kernel<<<gq, 256, 0, stream>>>(
            y16, DD, /*a_off_z=*/HD, WqT + (long)l * HD * HD, HD, 0,
            bq + (long)l * HD, nullptr, q16, DD, /*c_off_z=*/HD, MM, HD, HD, 0);
        gemm_f16_kernel<<<gq, 256, 0, stream>>>(
            y16, DD, HD, WkT + (long)l * HD * HD, HD, 0,
            bk + (long)l * HD, nullptr, k16, DD, HD, MM, HD, HD, 0);
        gemm_f16_kernel<<<gq, 256, 0, stream>>>(
            y16, DD, HD, WvT + (long)l * HD * HD, HD, 0,
            bv + (long)l * HD, nullptr, v16, DD, HD, MM, HD, HD, 0);

        // fused causal attention
        attn_flash_kernel<<<dim3(SS / 128, BB * HH), 256, 0, stream>>>(
            q16, k16, v16, ao16, scale);

        // attn output projection: M=2048, N=1024, K=1024
        gemm_f16_kernel<<<dim3(MM / 64, DD / 64, 1), 256, 0, stream>>>(
            ao16, DD, 0, WfcT + (long)l * DD * DD, DD, 0,
            bfc + (long)l * DD, tmpf, nullptr, DD, 0, MM, DD, DD, 0);

        // add = LN(o + y)
        ln_res_kernel<<<MM, 256, 0, stream>>>(
            tmpf, y_f32, ln1_w + (long)l * DD, ln1_b + (long)l * DD, addf, add16);

        // ff = GELU(add @ W1 + b1): M=2048, N=4096, K=1024
        gemm_f16_kernel<<<dim3(MM / 64, DFF / 64, 1), 256, 0, stream>>>(
            add16, DD, 0, W1T + (long)l * DD * DFF, DD, 0,
            b1 + (long)l * DFF, nullptr, ff16, DFF, 0, MM, DFF, DD, 1);

        // ff @ W2 + b2: M=2048, N=1024, K=4096
        gemm_f16_kernel<<<dim3(MM / 64, DD / 64, 1), 256, 0, stream>>>(
            ff16, DFF, 0, W2T + (long)l * DFF * DD, DFF, 0,
            b2 + (long)l * DD, tmpf, nullptr, DD, 0, MM, DD, DFF, 0);

        // y = LN(ff + add)
        ln_res_kernel<<<MM, 256, 0, stream>>>(
            tmpf, addf, ln2_w + (long)l * DD, ln2_b + (long)l * DD, y_f32, y16);
    }

    // final LN -> d_out (fp32)
    ln_res_kernel<<<MM, 256, 0, stream>>>(
        y_f32, nullptr, lnf_w, lnf_b, (float*)d_out, nullptr);

    (void)in_sizes; (void)n_in; (void)out_size; (void)ws_size;
}
